// VectorQuantizer_44530220925010
// MI455X (gfx1250) — matrix operand order repaired and checked by reading, not compiled, not run on hardware
//
#include <hip/hip_runtime.h>
#include <hip/hip_bf16.h>
#include <stdint.h>

// ---------------------------------------------------------------------------
// VQ-VAE vector quantizer for MI455X (gfx1250, wave32, WMMA bf16 + TDM).
// N=32768 rows x D=512 vs K=4096 codes. argmin(||x||^2 - 2x.c + ||c||^2)
// == argmin(||c||^2 - 2 x.c)  -> no x_sq, never materialize NxK.
// Codebook is pre-swizzled to WMMA B-fragment layout in global memory; the
// Tensor Data Mover DMA-stages one 16KB code tile per block into LDS
// (double-buffered, TENSORcnt-synced) and all 4 waves feed WMMAs from LDS.
// sched_group_barrier directives force: 16x ds_load clause -> (wmma + 2 ds)x8
// -> 8x wmma, so LDS latency hides behind WMMA execution.
// ---------------------------------------------------------------------------

typedef __attribute__((ext_vector_type(16))) __bf16 v16bf;
typedef __attribute__((ext_vector_type(8)))  float  v8f;
typedef __attribute__((ext_vector_type(4)))  unsigned int u32x4;
typedef __attribute__((ext_vector_type(8)))  int  i32x8;
typedef __attribute__((ext_vector_type(4)))  int  i32x4;

#define D_DIM          512
#define CHUNKS         16          // 512 / 32 (K-dim per wmma)
#define TILE_CODES     16
#define ROWS_PER_WAVE  16
#define WAVES_PER_BLK  4
#define ROWS_PER_BLK   64
#define TILE_BYTES     (TILE_CODES * D_DIM * 2)   // 16384 B of bf16 per code tile

// sched_group_barrier masks
#define SG_WMMA   0x008
#define SG_DSREAD 0x100

static __device__ __forceinline__ unsigned short f32_bf16(float f) {
  union { float f; unsigned u; } v; v.f = f;
  unsigned x = v.u;
  return (unsigned short)((x + 0x7FFFu + ((x >> 16) & 1u)) >> 16); // RNE
}

// ---------------------------------------------------------------------------
// Kernel 1: codebook fp32 -> bf16 pre-swizzled into WMMA B-fragment layout,
// plus ||c||^2 and the codebook_values output copy.
// B layout used here (16-bit 32x16, contiguous k, not ISA table order): col N = lane%16; lanes 0-15 hold
// k 0..15 of the 32-chunk, lanes 16-31 hold k 16..31; 2 bf16 per VGPR.
// Fragment of lane L for tile t, chunk j lives at byte
//   t*16384 + j*1024 + L*32   (32 contiguous bytes per lane).
// ---------------------------------------------------------------------------
__global__ void cvt_codebook(const float* __restrict__ cb,
                             unsigned char* __restrict__ cb_swz,
                             float* __restrict__ c_sq,
                             float* __restrict__ cb_out) {
  const int c   = blockIdx.x;        // code index
  const int tid = threadIdx.x;       // 128 threads, 4 elems each
  const int k0  = tid * 4;

  const float4 v = *(const float4*)(cb + (size_t)c * D_DIM + k0);
  *(float4*)(cb_out + (size_t)c * D_DIM + k0) = v;   // codebook_values output

  float ss = v.x * v.x + v.y * v.y + v.z * v.z + v.w * v.w;

  unsigned lo = (unsigned)f32_bf16(v.x) | ((unsigned)f32_bf16(v.y) << 16);
  unsigned hi = (unsigned)f32_bf16(v.z) | ((unsigned)f32_bf16(v.w) << 16);

  const int t = c >> 4, Lm = c & 15;
  const int j = k0 >> 5, r = k0 & 31, half = r >> 4, e = r & 15;
  const int lane = Lm + half * 16;
  const size_t off = (size_t)t * TILE_BYTES + (size_t)j * 1024 + lane * 32 + e * 2;
  uint2 p; p.x = lo; p.y = hi;
  *(uint2*)(cb_swz + off) = p;

  // block reduce ||c||^2
  for (int s = 16; s; s >>= 1) ss += __shfl_xor(ss, s, 32);
  __shared__ float red[4];
  if ((tid & 31) == 0) red[tid >> 5] = ss;
  __syncthreads();
  if (tid == 0) c_sq[c] = red[0] + red[1] + red[2] + red[3];
}

// ---------------------------------------------------------------------------
// TDM: DMA one pre-swizzled 16KB code tile (4096 x 4-byte units, single line)
// from global to LDS. D# per ISA 8.3/8.4: group0 = {count/type/addr},
// group1 = {data_size=4B, tensor_dim0=4096, tensor_dim1=1, tile_dim0=4096,
// tile_dim1=1, stride=4096}. Groups 2/3 zero (<=2D tensor).
// ---------------------------------------------------------------------------
static __device__ __forceinline__ void tdm_load_tile(const unsigned char* gsrc,
                                                     unsigned lds_off) {
  const unsigned long long ga = (unsigned long long)(uintptr_t)gsrc;
  u32x4 g0;
  g0[0] = 1u;                                         // count=1, user mode
  g0[1] = lds_off;                                    // lds_addr (bytes)
  g0[2] = (unsigned)(ga & 0xFFFFFFFFu);               // global_addr[31:0]
  g0[3] = (unsigned)((ga >> 32) & 0x01FFFFFFu) | (2u << 30); // addr[56:32]|type=2
  i32x8 g1;
  g1[0] = (int)(2u << 16);        // workgroup_mask=0, data_size=4B, no pad/iter
  g1[1] = (int)(4096u << 16);     // tensor_dim0[15:0]=4096 (bits 63:48)
  g1[2] = (int)(1u << 16);        // tensor_dim0 hi=0, tensor_dim1=1
  g1[3] = (int)(4096u << 16);     // tensor_dim1 hi=0, tile_dim0=4096
  g1[4] = 1;                      // tile_dim1=1, tile_dim2=0
  g1[5] = 4096;                   // tensor_dim0_stride[31:0]
  g1[6] = 0;                      // stride hi / tensor_dim1_stride lo
  g1[7] = 0;
  const i32x4 z4 = {0, 0, 0, 0};
#if defined(__clang_major__) && (__clang_major__ >= 23)
  const i32x8 z8 = {0, 0, 0, 0, 0, 0, 0, 0};
  __builtin_amdgcn_tensor_load_to_lds(g0, g1, z4, z4, z8, 0);
#else
  __builtin_amdgcn_tensor_load_to_lds(g0, g1, z4, z4, 0);
#endif
}

// ---------------------------------------------------------------------------
// Helpers for the distance kernel.
// ---------------------------------------------------------------------------
static __device__ __forceinline__ void stage_rows(const float* __restrict__ x,
                                                  unsigned char* __restrict__ aLds,
                                                  int rowBase, int lane) {
  // 16 rows x 512 fp32 -> bf16 row-major in LDS (coalesced reads).
  for (int r = 0; r < 16; ++r) {
    const float* src = x + (size_t)(rowBase + r) * D_DIM;
    unsigned char* dst = aLds + r * (D_DIM * 2);
    for (int c0 = lane * 2; c0 < D_DIM; c0 += 64) {
      float2 v = *(const float2*)(src + c0);
      unsigned p = (unsigned)f32_bf16(v.x) | ((unsigned)f32_bf16(v.y) << 16);
      *(unsigned*)(dst + c0 * 2) = p;
    }
  }
}

static __device__ __forceinline__ void read_afrags(const unsigned char* __restrict__ aLds,
                                                   int mrow, int half, v16bf* aF) {
  // A 16x32 layout matching B: lane<16 row=mrow gets k {d..d+15};
  // lanes 16-31 get k {d+16..d+31}.
  const unsigned char* rowp = aLds + mrow * (D_DIM * 2);
#pragma unroll
  for (int j = 0; j < CHUNKS; ++j) {
    const int d = j * 32;
    union { v16bf v; uint4 q[2]; } u;
    u.q[0] = *(const uint4*)(rowp + (d + half * 16) * 2);
    u.q[1] = *(const uint4*)(rowp + (d + half * 16 + 8) * 2);
    aF[j] = u.v;
  }
}

// ---------------------------------------------------------------------------
// Kernel 2: fused distance + argmin. 4 waves x 16 rows; A resident in 128
// VGPRs/lane; B tiles TDM-staged into a 2x16KB LDS double buffer shared by
// all waves; scheduler directives pipeline ds_load clauses under the WMMAs.
// ---------------------------------------------------------------------------
__global__ __launch_bounds__(128) void vq_dist_argmin(
    const float* __restrict__ x,
    const unsigned char* __restrict__ cb_swz,
    const float* __restrict__ c_sq,
    int* __restrict__ nn_ws,
    float* __restrict__ nn_out,
    int K) {
  // [0,32KB): B double buffer (steady state). Whole 64KB used once for A stage.
  __shared__ unsigned char smem[WAVES_PER_BLK * 16 * D_DIM * 2]; // 64 KB

  const int tid  = threadIdx.x;
  const int wave = tid >> 5;
  const int lane = tid & 31;
  const int half = lane >> 4;   // 0: lanes 0-15, 1: lanes 16-31
  const int mrow = lane & 15;
  const int rowBase = blockIdx.x * ROWS_PER_BLK + wave * ROWS_PER_WAVE;

  // Phase 1: stage A (16 rows/wave) through per-wave 16KB LDS region, then
  // pull the WMMA A fragments into registers (128 VGPRs/lane).
  v16bf aF[CHUNKS];
  unsigned char* aLds = smem + wave * (16 * D_DIM * 2);
  stage_rows(x, aLds, rowBase, lane);
  __syncthreads();
  read_afrags(aLds, mrow, half, aF);
  __syncthreads();            // everyone done with LDS before TDM reuses it

  // Generic shared pointer -> 32-bit LDS offset (addr[31:0], ISA 10.2).
  const unsigned ldsBase = (unsigned)(uintptr_t)(&smem[0]);

  float best[8]; int bidx[8];
#pragma unroll
  for (int i = 0; i < 8; ++i) { best[i] = 3.4e38f; bidx[i] = 0; }

  const int nTiles = K / TILE_CODES;

  // Preload tile 0 via the Tensor Data Mover.
  if (wave == 0) {
    tdm_load_tile(cb_swz, ldsBase);
    __builtin_amdgcn_s_wait_tensorcnt(0);
  }
  __syncthreads();

  for (int t = 0; t < nTiles; ++t) {
    const unsigned char* bb = smem + (t & 1) * TILE_BYTES;
    // Kick the DMA for tile t+1 into the other buffer, then compute tile t.
    if (wave == 0 && (t + 1) < nTiles)
      tdm_load_tile(cb_swz + (size_t)(t + 1) * TILE_BYTES,
                    ldsBase + (unsigned)(((t + 1) & 1) * TILE_BYTES));

    const int code = t * TILE_CODES + mrow;
    const float csq = c_sq[code];     // hoisted: overlaps with the MAC loop

    v8f acc0 = {}, acc1 = {};
    {
      uint4 b0[16], b1[16];
      // Batch 0: 8 fragments (16x ds_load_b128).
#pragma unroll
      for (int j = 0; j < 8; ++j) {
        const unsigned char* p = bb + (size_t)j * 1024 + lane * 32;
        b0[2 * j]     = *(const uint4*)(p);
        b0[2 * j + 1] = *(const uint4*)(p + 16);
      }
      // Batch 1: 8 fragments (scheduled under batch-0 WMMAs by directives).
#pragma unroll
      for (int j = 0; j < 8; ++j) {
        const unsigned char* p = bb + (size_t)(8 + j) * 1024 + lane * 32;
        b1[2 * j]     = *(const uint4*)(p);
        b1[2 * j + 1] = *(const uint4*)(p + 16);
      }
      // Batch 0 WMMAs.
#pragma unroll
      for (int j = 0; j < 8; ++j) {
        union { v16bf v; uint4 q[2]; } u;
        u.q[0] = b0[2 * j]; u.q[1] = b0[2 * j + 1];
        if ((j & 1) == 0)
          acc0 = __builtin_amdgcn_wmma_f32_16x16x32_bf16(false, aF[j], false, u.v,
                                                         (short)0, acc0, false, false);
        else
          acc1 = __builtin_amdgcn_wmma_f32_16x16x32_bf16(false, aF[j], false, u.v,
                                                         (short)0, acc1, false, false);
      }
      // Batch 1 WMMAs.
#pragma unroll
      for (int j = 0; j < 8; ++j) {
        union { v16bf v; uint4 q[2]; } u;
        u.q[0] = b1[2 * j]; u.q[1] = b1[2 * j + 1];
        if ((j & 1) == 0)
          acc0 = __builtin_amdgcn_wmma_f32_16x16x32_bf16(false, aF[8 + j], false, u.v,
                                                         (short)0, acc0, false, false);
        else
          acc1 = __builtin_amdgcn_wmma_f32_16x16x32_bf16(false, aF[8 + j], false, u.v,
                                                         (short)0, acc1, false, false);
      }
      // Enforce the software pipeline:
      //   16 DS reads ; (1 WMMA ; 2 DS reads) x8 ; 8 WMMAs
      __builtin_amdgcn_sched_group_barrier(SG_DSREAD, 16, 0);
#pragma unroll
      for (int j = 0; j < 8; ++j) {
        __builtin_amdgcn_sched_group_barrier(SG_WMMA, 1, 0);
        __builtin_amdgcn_sched_group_barrier(SG_DSREAD, 2, 0);
      }
      __builtin_amdgcn_sched_group_barrier(SG_WMMA, 8, 0);
    }

#pragma unroll
    for (int i = 0; i < 8; ++i) {
      float dist = csq - 2.0f * (acc0[i] + acc1[i]);
      if (dist < best[i]) { best[i] = dist; bidx[i] = code; }
    }

    if (wave == 0) __builtin_amdgcn_s_wait_tensorcnt(0);
    __syncthreads();          // tile t+1 resident; all waves done with tile t
  }

  // Cross-lane argmin inside each 16-lane half (ties -> lowest index).
#pragma unroll
  for (int i = 0; i < 8; ++i) {
    float bv = best[i]; int bi = bidx[i];
    for (int s = 8; s >= 1; s >>= 1) {
      float ov = __shfl_xor(bv, s, 16);
      int   oi = __shfl_xor(bi, s, 16);
      if (ov < bv || (ov == bv && oi < bi)) { bv = ov; bi = oi; }
    }
    if (mrow == 0) {                       // lanes 0 (rows 0-7) and 16 (rows 8-15)
      const int row = rowBase + i + half * 8;
      nn_ws[row]  = bi;
      nn_out[row] = (float)bi;
    }
  }
}

// ---------------------------------------------------------------------------
// Kernel 3: gather centroid, quantized_ste = q, loss = 1.25*(q-x)^2, counts.
// ---------------------------------------------------------------------------
__global__ void vq_gather(const float* __restrict__ x,
                          const float* __restrict__ cb,
                          const int* __restrict__ nn,
                          float* __restrict__ q_out,
                          float* __restrict__ loss_out,
                          float* __restrict__ counts) {
  const int row = blockIdx.x;
  const int tid = threadIdx.x;          // 256 threads, 2 elems each
  const int idx = nn[row];
  if (tid == 0) atomicAdd(counts + idx, 1.0f);

  const size_t o = (size_t)row * D_DIM + tid * 2;
  const float2 xv = *(const float2*)(x + o);
  const float2 cv = *(const float2*)(cb + (size_t)idx * D_DIM + tid * 2);
  const float dx = cv.x - xv.x, dy = cv.y - xv.y;
  float2 l; l.x = 1.25f * dx * dx; l.y = 1.25f * dy * dy;
  *(float2*)(q_out + o)    = cv;
  *(float2*)(loss_out + o) = l;
}

__global__ void vq_zero(float* __restrict__ p, int n) {
  int i = blockIdx.x * blockDim.x + threadIdx.x;
  if (i < n) p[i] = 0.0f;
}

__global__ void vq_ema(const float* __restrict__ old_c,
                       const float* __restrict__ counts,
                       float* __restrict__ out, int n) {
  int i = blockIdx.x * blockDim.x + threadIdx.x;
  if (i < n) out[i] = 0.99f * old_c[i] + 0.01f * counts[i];
}

// ---------------------------------------------------------------------------
extern "C" void kernel_launch(void* const* d_in, const int* in_sizes, int n_in,
                              void* d_out, int out_size, void* d_ws, size_t ws_size,
                              hipStream_t stream) {
  const float* x          = (const float*)d_in[0];   // [N, 512]
  const float* cb         = (const float*)d_in[1];   // [K, 512]
  const float* old_counts = (const float*)d_in[2];   // [K]
  const int N = in_sizes[0] / D_DIM;                 // 32768
  const int K = in_sizes[2];                         // 4096

  float* out = (float*)d_out;
  size_t off = 0;
  float* q_out    = out + off; off += (size_t)N * D_DIM;   // quantized_ste
  float* loss_out = out + off; off += (size_t)N * D_DIM;   // quant_loss
  float* nn_out   = out + off; off += (size_t)N;           // nn_idx (as f32)
  float* cb_out   = out + off; off += (size_t)K * D_DIM;   // codebook_values
  float* cnt_out  = out + off;                             // new_cluster_counts

  unsigned char* ws = (unsigned char*)d_ws;
  unsigned char* cb_swz = ws;                                 // K*512*2 B bf16
  float* c_sq   = (float*)(ws + (size_t)K * D_DIM * 2);       // [K]
  float* counts = c_sq + K;                                   // [K]
  int*   nn_ws  = (int*)(counts + K);                         // [N]

  cvt_codebook<<<K, 128, 0, stream>>>(cb, cb_swz, c_sq, cb_out);
  vq_zero<<<(K + 255) / 256, 256, 0, stream>>>(counts, K);
  vq_dist_argmin<<<N / ROWS_PER_BLK, 128, 0, stream>>>(x, cb_swz, c_sq,
                                                       nn_ws, nn_out, K);
  vq_gather<<<N, 256, 0, stream>>>(x, cb, nn_ws, q_out, loss_out, counts);
  vq_ema<<<(K + 255) / 256, 256, 0, stream>>>(old_counts, counts, cnt_out, K);
}
